// BiSRNNFMNIST_20538533609762
// MI455X (gfx1250) — matrix operand order repaired and checked by reading, not compiled, not run on hardware
//
#include <hip/hip_runtime.h>

typedef __attribute__((ext_vector_type(16))) _Float16 v16h;
typedef __attribute__((ext_vector_type(8)))  _Float16 v8h;
typedef __attribute__((ext_vector_type(8)))  float    v8f;

#if __has_builtin(__builtin_amdgcn_sched_barrier)
#define SCHED_FENCE() __builtin_amdgcn_sched_barrier(0)
#else
#define SCHED_FENCE() asm volatile("" ::: "memory")
#endif

#define BSZ 128
#define T_STEPS 784
#define H 200
#define HP 224            // K padded to 7*32
#define NT 13             // N tiles (13*16 = 208 >= 200)
#define KT 7              // K tiles
#define TILE_HALF 512     // halves per packed B tile (32*16)
#define NTILES (KT*NT)    // 91 tiles per direction
#define WSB_HALFS (2*NTILES*TILE_HALF)   // 93184 halves
#define WSV_BYTE_OFF 186368              // bytes: after B tiles (4-aligned)

// Pack Wr^T (B-operand of s @ Wr.T) into WMMA-B fragment layout:
// tile(dir,kt,nt): entry [lane*16 + s] = Wr[n, k], n_local = lane%16, k_local = table order (h=lane/16), zero-padded.
// Rows k >= 200 are zero, which also annihilates any junk spikes written
// into the padded LDS columns (lets the scan kernel skip store guards).
__global__ void pack_wr_kernel(const float* __restrict__ Wr_fw,
                               const float* __restrict__ Wr_bw,
                               _Float16* __restrict__ wsB) {
    int id = blockIdx.x * blockDim.x + threadIdx.x;
    if (id >= WSB_HALFS) return;
    int dir  = id / (NTILES * TILE_HALF);
    int r    = id % (NTILES * TILE_HALF);
    int tile = r >> 9;            // 0..90
    int kt   = tile / NT;
    int nt   = tile % NT;
    int kl   = ((r >> 8) & 1) * 8 + (r & 7) + ((r >> 3) & 1) * 16;
    int nl   = (r >> 4) & 15;
    int k = kt * 32 + kl;
    int n = nt * 16 + nl;
    const float* Wr = dir ? Wr_bw : Wr_fw;
    float v = (k < H && n < H) ? Wr[n * H + k] : 0.0f;
    wsB[id] = (_Float16)v;
}

// v_dir[h] = sum_c w1[c] * Wd_dir[h,c]   (input GEMM collapsed since D==1)
__global__ void pack_v_kernel(const float* __restrict__ w1,
                              const float* __restrict__ Wd_fw,
                              const float* __restrict__ Wd_bw,
                              float* __restrict__ wsV) {
    int dir = blockIdx.x;
    int h = threadIdx.x;
    if (h >= 208) return;
    const float* Wd = dir ? Wd_bw : Wd_fw;
    float s = 0.0f;
    if (h < H) {
        for (int c = 0; c < 784; ++c) s += w1[c] * Wd[h * 784 + c];
    }
    wsV[dir * 208 + h] = s;
}

// One workgroup = 16 batch rows x one direction; 13 waves, one 16x16 N-tile each.
// Wr^T fragments resident in VGPRs for all 784 steps; spikes double-buffered in
// LDS (f16, exact for 0/1); x slice resident in LDS; one DS-only barrier per step.
__global__ __launch_bounds__(416) void bisrnn_scan_kernel(
    const float* __restrict__ x,
    const float* __restrict__ mem0_fw,
    const float* __restrict__ mem0_bw,
    const _Float16* __restrict__ wsB,
    const float* __restrict__ wsV,
    float* __restrict__ out) {

    __shared__ _Float16 sfh[2 * 16 * HP];     // double-buffered spike matrix (14 KB)
    __shared__ float    xs[16 * T_STEPS];     // this block's x slice (49 KB)

    const int dir  = blockIdx.x & 1;
    const int b0   = (blockIdx.x >> 1) * 16;
    const int wv   = threadIdx.x >> 5;
    const int lane = threadIdx.x & 31;
    const int ln   = lane & 15;
    const int hig  = lane >> 4;
    const int nt   = wv;                // this wave's N tile
    const int n    = nt * 16 + ln;      // global hidden index (C/D layout: N=lane)

    // ---- preload B fragments: lane L holds n=L%16, halves hold k in table order (h=L/16) (contig 32B)
    v16h bfrag[KT];
    {
        const _Float16* bp = wsB + ((size_t)dir * NTILES + nt) * TILE_HALF + lane * 16;
        #pragma unroll
        for (int kt = 0; kt < KT; ++kt)
            bfrag[kt] = *(const v16h*)(bp + (size_t)kt * NT * TILE_HALF);
    }
    const float vn = wsV[dir * 208 + n];
    const float* mem0 = dir ? mem0_bw : mem0_fw;

    // ---- persistent per-element state in WMMA C/D layout: m = 8*hig + i
    v8f memv, badv, sprev;
    #pragma unroll
    for (int i = 0; i < 8; ++i) {
        int m = 8 * hig + i;
        memv[i]  = (n < H) ? mem0[(b0 + m) * H + n] : 0.0f;
        badv[i]  = 0.01f;    // b0 = b_j0
        sprev[i] = 0.0f;
    }

    for (int idx = threadIdx.x; idx < 2 * 16 * HP; idx += blockDim.x)
        sfh[idx] = (_Float16)0.0f;
    for (int idx = threadIdx.x; idx < 16 * T_STEPS; idx += blockDim.x) {
        int m = idx / T_STEPS, c = idx % T_STEPS;
        xs[idx] = x[(b0 + m) * T_STEPS + c];
    }
    __syncthreads();

    // ---- per-lane output base pointers, advanced +-400 floats per step so all
    //      8 stores per step use immediate offsets (max 3.76 MB < 24-bit IOFFSET)
    const size_t t0 = dir ? (size_t)(T_STEPS - 1) : 0;
    float* op0 = out + ((size_t)(b0 + 8 * hig) * T_STEPS + t0) * 400 + (size_t)dir * H + n;
    float* op1 = out + ((size_t)(b0 + 8 * hig + 4) * T_STEPS + t0) * 400 + (size_t)dir * H + n;
    const ptrdiff_t dstep = dir ? -400 : 400;

    const float alpha = 0.95122945f;   // exp(-1/20)
    const float ro    = 0.99501246f;   // exp(-1/200)
    const float oma   = 1.0f - alpha;
    const float omr   = 1.0f - ro;

    const float* xbase = xs + (8 * hig) * T_STEPS;

    // One scan step with compile-time buffer parity P (read buf P, write buf P^1).
    auto step = [&](int t, int P) {
        const int xin_t = dir ? (t == 0 ? 0 : T_STEPS - t) : t;   // (-t) % T

        // ---- issue ALL 22 LDS loads (8x x, 14x A-frag) before any WMMA; the
        //      sched_barrier pins them, so dscnt waits become partial/pipelined
        const float* xr = xbase + xin_t;
        float xv[8];
        #pragma unroll
        for (int i = 0; i < 8; ++i) xv[i] = xr[i * T_STEPS];

        const _Float16* ar = sfh + P * (16 * HP) + ln * HP + 8 * hig;
        v16h afr[KT];
        #pragma unroll
        for (int kt = 0; kt < KT; ++kt) {
            v8h a0 = *(const v8h*)(ar + kt * 32);
            v8h a1 = *(const v8h*)(ar + kt * 32 + 16);
            afr[kt] = __builtin_shufflevector(a0, a1,
                        0,1,2,3,4,5,6,7,8,9,10,11,12,13,14,15);
        }
        SCHED_FENCE();

        // ---- recurrent GEMM: acc[m][n] = sum_k S[m][k] * Wr[n][k], 2 chains
        v8f acc0 = {0,0,0,0,0,0,0,0};
        v8f acc1 = {0,0,0,0,0,0,0,0};
        #pragma unroll
        for (int kt = 0; kt < KT; ++kt) {
            if (kt & 1)
                acc1 = __builtin_amdgcn_wmma_f32_16x16x32_f16(
                         false, afr[kt], false, bfrag[kt], (short)0, acc1, false, false);
            else
                acc0 = __builtin_amdgcn_wmma_f32_16x16x32_f16(
                         false, afr[kt], false, bfrag[kt], (short)0, acc0, false, false);
        }

        // ---- adaptive-LIF update; spikes into buffer P^1 (no guard: padded
        //      columns hit zeroed B rows, so junk there is annihilated)
        _Float16* sw = sfh + (P ^ 1) * (16 * HP) + (8 * hig) * HP + n;
        #pragma unroll
        for (int i = 0; i < 8; ++i) {
            float d    = acc0[i] + acc1[i] + xv[i] * vn;
            float bnew = ro * badv[i] + omr * sprev[i];
            float Bth  = 0.01f + 1.8f * bnew;
            float mnew = memv[i] * alpha + oma * d - Bth * sprev[i];
            float snew = (mnew - Bth) > 0.0f ? 1.0f : 0.0f;
            memv[i] = mnew; badv[i] = bnew; sprev[i] = snew;
            sw[i * HP] = (_Float16)snew;
        }

        // ---- DS-only split barrier: wait LDS traffic, signal arrival, then
        //      issue the fire-and-forget NT global stores under barrier skew,
        //      then wait. (No STOREcnt drain per step, unlike __syncthreads.)
        asm volatile("s_wait_dscnt 0x0\n\ts_barrier_signal -1" ::: "memory");
        if (n < H) {
            #pragma unroll
            for (int i = 0; i < 4; ++i) {
                __builtin_nontemporal_store(sprev[i],     op0 + (size_t)i * T_STEPS * 400);
                __builtin_nontemporal_store(sprev[i + 4], op1 + (size_t)i * T_STEPS * 400);
            }
        }
        op0 += dstep; op1 += dstep;
        asm volatile("s_barrier_wait -1" ::: "memory");
    };

    #pragma unroll 1
    for (int t = 0; t < T_STEPS; t += 2) {   // T_STEPS is even
        step(t,     0);
        step(t + 1, 1);
    }
}

extern "C" void kernel_launch(void* const* d_in, const int* in_sizes, int n_in,
                              void* d_out, int out_size, void* d_ws, size_t ws_size,
                              hipStream_t stream) {
    const float* x       = (const float*)d_in[0];
    const float* w1      = (const float*)d_in[1];
    const float* Wd_fw   = (const float*)d_in[2];
    const float* Wr_fw   = (const float*)d_in[3];
    const float* Wd_bw   = (const float*)d_in[4];
    const float* Wr_bw   = (const float*)d_in[5];
    const float* mem0_fw = (const float*)d_in[6];
    const float* mem0_bw = (const float*)d_in[7];
    float* out = (float*)d_out;

    _Float16* wsB = (_Float16*)d_ws;
    float*    wsV = (float*)((char*)d_ws + WSV_BYTE_OFF);

    pack_wr_kernel<<<(WSB_HALFS + 255) / 256, 256, 0, stream>>>(Wr_fw, Wr_bw, wsB);
    pack_v_kernel<<<2, 256, 0, stream>>>(w1, Wd_fw, Wd_bw, wsV);
    // 8 batch tiles x 2 directions; 13 waves each
    bisrnn_scan_kernel<<<16, 416, 0, stream>>>(x, mem0_fw, mem0_bw, wsB, wsV, out);
}
